// MultiHeadAttention_90666759618568
// MI455X (gfx1250) — matrix operand order repaired and checked by reading, not compiled, not run on hardware
//
#include <hip/hip_runtime.h>
#include <hip/hip_bf16.h>
#include <stdint.h>

// ---------------------------------------------------------------------------
// MultiHeadAttention for MI455X (gfx1250, wave32, WMMA + async LDS staging)
//   B=4, H=8, S=2048, D=128, EMB=1024
//   d_out = [ x : 4*2048*1024 f32 | attention : 4*8*2048*2048 f32 ]
//
// Pipeline:
//   proj Q,K  : f32 GEMM-> bf16 [B,H,S,D]      (sync staging, f32->bf16 cvt)
//   proj V    : f32 GEMM-> bf16 [B,H,D,S] (T)  (transposed epilogue store)
//   scores    : bf16 QK^T -> f32 attn region   (fully async, double-buffered)
//   softmax   : in-place f32 rows
//   attn @ V  : A f32 sync-cvt, B bf16 async   (reads 537MB attn: HBM-bound)
//   out proj  : A bf16 async, B f32 sync-cvt -> f32 x region
// ---------------------------------------------------------------------------

typedef __attribute__((ext_vector_type(16))) __bf16 v16bf;
typedef __attribute__((ext_vector_type(8)))  float  v8f;

#define LDSTR 40   // padded LDS row stride in bf16 elements (80B, 16B-aligned)

union FragBF {
    v16bf v;
    uint4 q[2];
};

union Pack4 {
    __bf16 h[4];
    uint2  u;
};

// ---------------------------------------------------------------------------
// Fragment load + 8x WMMA
// ---------------------------------------------------------------------------
__device__ __forceinline__ void wmma_step(const __bf16* sA, const __bf16* sB,
                                          v8f (&acc)[2][4],
                                          int wm, int wn, int lrow, int khalf)
{
    // A 16x32 layout: lane<16 -> K {0..7,16..23}; lane>=16 -> K {8..15,24..31}
    FragBF fa[2];
#pragma unroll
    for (int mt = 0; mt < 2; ++mt) {
        const __bf16* p = &sA[(wm * 32 + mt * 16 + lrow) * LDSTR + khalf * 8];
        fa[mt].q[0] = *(const uint4*)p;
        fa[mt].q[1] = *(const uint4*)(p + 16);
    }
    // B 32x16 layout: lane<16 -> K {0..7,16..23}; lane>=16 -> K {8..15,24..31}, col = lane&15
    FragBF fb[4];
#pragma unroll
    for (int nt = 0; nt < 4; ++nt) {
        const __bf16* p = &sB[(wn * 64 + nt * 16 + lrow) * LDSTR + khalf * 8];
        fb[nt].q[0] = *(const uint4*)p;
        fb[nt].q[1] = *(const uint4*)(p + 16);
    }
#pragma unroll
    for (int mt = 0; mt < 2; ++mt)
#pragma unroll
        for (int nt = 0; nt < 4; ++nt)
            acc[mt][nt] = __builtin_amdgcn_wmma_f32_16x16x32_bf16(
                false, fa[mt].v, false, fb[nt].v,
                (short)0, acc[mt][nt], false, false);
}

// C/D 16x16 layout: VGPR v, lane<16 -> (M=v, N=lane); lane>=16 -> (M=v+8, N=lane-16)
// C_TRANS: store C[n*ldc + m] instead of C[m*ldc + n] (transposed output)
template<bool C_BF16, bool C_TRANS, bool HAS_BIAS>
__device__ __forceinline__ void store_tile(v8f (&acc)[2][4], void* Cp, int ldc,
                                           const float* bias, float scale,
                                           int wm, int wn, int lrow, int khalf)
{
#pragma unroll
    for (int mt = 0; mt < 2; ++mt) {
#pragma unroll
        for (int nt = 0; nt < 4; ++nt) {
            const int nl = wn * 64 + nt * 16 + lrow;
            float bv = 0.0f;
            if (HAS_BIAS) bv = bias[nl];
            const int mb = wm * 32 + mt * 16 + khalf * 8;
#pragma unroll
            for (int v = 0; v < 8; ++v) {
                const float r = acc[mt][nt][v] * scale + bv;
                const size_t idx = C_TRANS ? ((size_t)nl * ldc + (mb + v))
                                           : ((size_t)(mb + v) * ldc + nl);
                if (C_BF16) ((__bf16*)Cp)[idx] = (__bf16)r;
                else        ((float*)Cp)[idx]  = r;
            }
        }
    }
}

// ---------------------------------------------------------------------------
// Sync staging (register round-trip, f32->bf16 convert via v_cvt_pk_bf16_f32)
// ---------------------------------------------------------------------------
template<bool SRC_BF16>
__device__ __forceinline__ void stage_rowmajor_32(const void* src, int ld, int kt,
                                                  __bf16* sm, int tid)
{
#pragma unroll
    for (int i = 0; i < 4; ++i) {
        const int e   = tid + i * 256;   // 0..1023
        const int row = e >> 3;          // 0..127
        const int kq  = (e & 7) << 2;    // 0,4,...,28
        Pack4 pk;
        if (SRC_BF16) {
            pk.u = *(const uint2*)((const __bf16*)src + (size_t)row * ld + kt * 32 + kq);
        } else {
            const float4 f = *(const float4*)((const float*)src + (size_t)row * ld + kt * 32 + kq);
            pk.h[0] = (__bf16)f.x; pk.h[1] = (__bf16)f.y;
            pk.h[2] = (__bf16)f.z; pk.h[3] = (__bf16)f.w;
        }
        *(uint2*)&sm[row * LDSTR + kq] = pk.u;
    }
}

// ---------------------------------------------------------------------------
// Async staging (bf16, row-major, K-contiguous): global_load_async_to_lds_b128
// ---------------------------------------------------------------------------
__device__ __forceinline__ void issue_async_tile(const __bf16* src, int ld, int kt,
                                                 __bf16* smbuf, int tid)
{
    // 128 rows x 32 bf16 = 512 chunks of 16B; 2 async b128 copies per thread
#pragma unroll
    for (int i = 0; i < 2; ++i) {
        const int c   = tid + i * 256;    // 0..511
        const int row = c >> 2;           // 0..127
        const int kq  = (c & 3) << 3;     // 0,8,16,24 bf16 elems
        const unsigned lds_addr =
            (unsigned)(uintptr_t)(smbuf + row * LDSTR + kq);
        const unsigned goff =
            (unsigned)(((unsigned)row * (unsigned)ld + (unsigned)(kt * 32 + kq)) * 2u);
        asm volatile("global_load_async_to_lds_b128 %0, %1, %2"
                     :: "v"(lds_addr), "v"(goff), "s"(src)
                     : "memory");
    }
}

// ---------------------------------------------------------------------------
// Generic 128x128 C-tile GEMM, per-operand sync/async staging.
//   A, B: row-major, K contiguous. *_ASYNC requires the operand to be bf16.
// ---------------------------------------------------------------------------
template<bool A_BF16, bool A_ASYNC, bool B_BF16, bool B_ASYNC,
         bool C_BF16, bool C_TRANS, bool HAS_BIAS>
__device__ __forceinline__ void gemm_tile_128x128(
    const void* Ap, int lda,
    const void* Bp, int ldb,
    int ksteps,
    void* Cp, int ldc,
    const float* bias,   // pre-offset to n0; indexed by n_local (0..127)
    float scale)
{
    __shared__ __bf16 smA[128 * LDSTR];
    __shared__ __bf16 smB[128 * LDSTR];

    const int tid   = threadIdx.x;
    const int lane  = tid & 31;
    const int wid   = tid >> 5;          // 8 waves
    const int wm    = wid & 3;           // 4 M quadrants of 32 rows
    const int wn    = wid >> 2;          // 2 N halves of 64 cols
    const int lrow  = lane & 15;
    const int khalf = lane >> 4;

    v8f acc[2][4];
#pragma unroll
    for (int i = 0; i < 2; ++i)
#pragma unroll
        for (int j = 0; j < 4; ++j)
#pragma unroll
            for (int e = 0; e < 8; ++e) acc[i][j][e] = 0.0f;

    const size_t aes = A_BF16 ? 2 : 4;
    const size_t bes = B_BF16 ? 2 : 4;

    for (int kt = 0; kt < ksteps; ++kt) {
        // ---- stage A tile (128 x 32) ----
        if (A_ASYNC) issue_async_tile((const __bf16*)Ap, lda, kt, smA, tid);
        else         stage_rowmajor_32<A_BF16>(Ap, lda, kt, smA, tid);

        // ---- stage B tile (128 x 32) ----
        if (B_ASYNC) issue_async_tile((const __bf16*)Bp, ldb, kt, smB, tid);
        else         stage_rowmajor_32<B_BF16>(Bp, ldb, kt, smB, tid);

        // ---- prefetch next K-tile of sync operands (global_prefetch_b8) ----
        if (kt + 1 < ksteps) {
            const int prow = tid >> 1;               // 0..127
            const int pkq  = (tid & 1) << 4;         // 0 or 16
            if (!A_ASYNC)
                __builtin_prefetch((const char*)Ap +
                                   aes * ((size_t)prow * lda + (kt + 1) * 32 + pkq), 0, 1);
            if (!B_ASYNC)
                __builtin_prefetch((const char*)Bp +
                                   bes * ((size_t)prow * ldb + (kt + 1) * 32 + pkq), 0, 1);
        }

        if (A_ASYNC || B_ASYNC)
            asm volatile("s_wait_asynccnt 0x0" ::: "memory");
        __syncthreads();

        wmma_step(smA, smB, acc, wm, wn, lrow, khalf);

        __syncthreads();
    }

    store_tile<C_BF16, C_TRANS, HAS_BIAS>(acc, Cp, ldc, bias, scale,
                                          wm, wn, lrow, khalf);
}

// ---------------------------------------------------------------------------
// Fully-async, double-buffered GEMM: A,B bf16 row-major, C f32 (scores).
// One barrier per K-step; next tiles stream into the other buffer under WMMA.
// ---------------------------------------------------------------------------
__device__ __forceinline__ void gemm_tile_async(
    const __bf16* Ap, int lda,
    const __bf16* Bp, int ldb,
    int ksteps,
    float* Cp, int ldc,
    float scale)
{
    __shared__ __bf16 smA[2][128 * LDSTR];
    __shared__ __bf16 smB[2][128 * LDSTR];

    const int tid   = threadIdx.x;
    const int lane  = tid & 31;
    const int wid   = tid >> 5;
    const int wm    = wid & 3;
    const int wn    = wid >> 2;
    const int lrow  = lane & 15;
    const int khalf = lane >> 4;

    v8f acc[2][4];
#pragma unroll
    for (int i = 0; i < 2; ++i)
#pragma unroll
        for (int j = 0; j < 4; ++j)
#pragma unroll
            for (int e = 0; e < 8; ++e) acc[i][j][e] = 0.0f;

    issue_async_tile(Ap, lda, 0, smA[0], tid);
    issue_async_tile(Bp, ldb, 0, smB[0], tid);

    for (int kt = 0; kt < ksteps; ++kt) {
        // this wave's copies into buf[kt&1] done, then rendezvous
        asm volatile("s_wait_asynccnt 0x0" ::: "memory");
        __syncthreads();

        const int cur = kt & 1;
        if (kt + 1 < ksteps) {
            // safe: buf[cur^1] was last read in iteration kt-1; every wave has
            // passed the barrier above, so nobody is still reading it
            issue_async_tile(Ap, lda, kt + 1, smA[cur ^ 1], tid);
            issue_async_tile(Bp, ldb, kt + 1, smB[cur ^ 1], tid);
        }

        wmma_step(smA[cur], smB[cur], acc, wm, wn, lrow, khalf);
    }

    store_tile<false, false, false>(acc, Cp, ldc, nullptr, scale,
                                    wm, wn, lrow, khalf);
}

// ---------------------------------------------------------------------------
// Kernels
// ---------------------------------------------------------------------------

// P = X @ W^T + b, bf16.
//   TRANS=false: head-split [B,H,S,D] (Q, K)
//   TRANS=true : head-split transposed [B,H,D,S] (V, so attn@V's B is row-major)
template<bool TRANS>
__global__ __launch_bounds__(256) void proj_kernel(
    const float* __restrict__ X, const float* __restrict__ W,
    const float* __restrict__ bias, __bf16* __restrict__ out)
{
    const int h  = blockIdx.x;              // n0 = h*128 (one head per N tile)
    const int m0 = blockIdx.y * 128;
    const int b  = m0 >> 11;                // / 2048
    const int s0 = m0 & 2047;
    __bf16* C;
    int ldc;
    if (TRANS) { C = out + (((size_t)b * 8 + h) * 128) * 2048 + s0; ldc = 2048; }
    else       { C = out + (((size_t)b * 8 + h) * 2048 + s0) * 128; ldc = 128;  }
    gemm_tile_128x128<false, false, false, false, true, TRANS, true>(
        X + (size_t)m0 * 1024, 1024,
        W + (size_t)h * 128 * 1024, 1024,
        32,
        C, ldc,
        bias + h * 128, 1.0f);
}

// scores = (Q @ K^T) / sqrt(128), f32 into attention region of d_out
__global__ __launch_bounds__(256) void scores_kernel(
    const __bf16* __restrict__ Q, const __bf16* __restrict__ Kmat,
    float* __restrict__ attn)
{
    const int bh = blockIdx.z;
    const int n0 = blockIdx.x * 128;
    const int m0 = blockIdx.y * 128;
    const __bf16* Ap = Q    + (size_t)bh * 2048 * 128 + (size_t)m0 * 128;
    const __bf16* Bp = Kmat + (size_t)bh * 2048 * 128 + (size_t)n0 * 128;
    float* C = attn + (size_t)bh * 2048 * 2048 + (size_t)m0 * 2048 + n0;
    gemm_tile_async(Ap, 128, Bp, 128, 4, C, 2048, 0.08838834764831845f);
}

// In-place row softmax over 2048-wide rows (one block per row)
__global__ __launch_bounds__(256) void softmax_kernel(float* __restrict__ attn)
{
    float* p = attn + (size_t)blockIdx.x * 2048;
    const int tid = threadIdx.x;
    const int lane = tid & 31, wid = tid >> 5;
    __shared__ float red[8];

    float v[8];
    float m = -3.0e38f;
#pragma unroll
    for (int i = 0; i < 8; ++i) { v[i] = p[tid + i * 256]; m = fmaxf(m, v[i]); }
#pragma unroll
    for (int o = 16; o > 0; o >>= 1) m = fmaxf(m, __shfl_xor(m, o));
    if (lane == 0) red[wid] = m;
    __syncthreads();
    if (tid < 8) {
        float t = red[tid];
        t = fmaxf(t, __shfl_xor(t, 4));
        t = fmaxf(t, __shfl_xor(t, 2));
        t = fmaxf(t, __shfl_xor(t, 1));
        if (tid == 0) red[0] = t;
    }
    __syncthreads();
    m = red[0];
    __syncthreads();

    float s = 0.0f;
#pragma unroll
    for (int i = 0; i < 8; ++i) { v[i] = __expf(v[i] - m); s += v[i]; }
#pragma unroll
    for (int o = 16; o > 0; o >>= 1) s += __shfl_xor(s, o);
    if (lane == 0) red[wid] = s;
    __syncthreads();
    if (tid < 8) {
        float t = red[tid];
        t += __shfl_xor(t, 4); t += __shfl_xor(t, 2); t += __shfl_xor(t, 1);
        if (tid == 0) red[0] = t;
    }
    __syncthreads();
    const float inv = 1.0f / red[0];
#pragma unroll
    for (int i = 0; i < 8; ++i) p[tid + i * 256] = v[i] * inv;
}

// ctx = attn @ V, bf16 merged-head store into [B,S,EMB]
// A = attention f32 (sync staging, f32->bf16 cvt); B = V^T bf16 (async staging)
__global__ __launch_bounds__(256) void attnv_kernel(
    const float* __restrict__ attn, const __bf16* __restrict__ Vt,
    __bf16* __restrict__ ctx)
{
    const int bh = blockIdx.z;
    const int m0 = blockIdx.y * 128;
    const int b = bh >> 3, h = bh & 7;
    const float*  Ap = attn + (size_t)bh * 2048 * 2048 + (size_t)m0 * 2048;
    const __bf16* Bp = Vt   + (size_t)bh * 128 * 2048;   // [N=D][K=S] row-major
    __bf16* C = ctx + ((size_t)b * 2048 + m0) * 1024 + (size_t)h * 128;
    gemm_tile_128x128<false, false, true, true, true, false, false>(
        Ap, 2048, Bp, 2048, 64, C, 1024, nullptr, 1.0f);
}

// x = ctx @ Wo^T + bo, f32 into x region of d_out
// A = ctx bf16 (async staging); B = Wo f32 (sync staging, f32->bf16 cvt)
__global__ __launch_bounds__(256) void out_kernel(
    const __bf16* __restrict__ ctx, const float* __restrict__ Wo,
    const float* __restrict__ bo, float* __restrict__ xout)
{
    const int n0 = blockIdx.x * 128;
    const int m0 = blockIdx.y * 128;
    gemm_tile_128x128<true, true, false, false, false, false, true>(
        ctx + (size_t)m0 * 1024, 1024,
        Wo + (size_t)n0 * 1024, 1024,
        32,
        xout + (size_t)m0 * 1024 + n0, 1024,
        bo + n0, 1.0f);
}

// ---------------------------------------------------------------------------
// Launch
// ---------------------------------------------------------------------------
extern "C" void kernel_launch(void* const* d_in, const int* in_sizes, int n_in,
                              void* d_out, int out_size, void* d_ws, size_t ws_size,
                              hipStream_t stream) {
    const float* query = (const float*)d_in[0];
    const float* key_  = (const float*)d_in[1];
    const float* value = (const float*)d_in[2];
    const float* Wq    = (const float*)d_in[3];
    const float* bq    = (const float*)d_in[4];
    const float* Wk    = (const float*)d_in[5];
    const float* bk    = (const float*)d_in[6];
    const float* Wv    = (const float*)d_in[7];
    const float* bv    = (const float*)d_in[8];
    const float* Wo    = (const float*)d_in[9];
    const float* bo    = (const float*)d_in[10];

    float* xout = (float*)d_out;                       // 4*2048*1024
    float* attn = xout + (size_t)4 * 2048 * 1024;      // 4*8*2048*2048

    const size_t QSZ = (size_t)4 * 8 * 2048 * 128;     // 8,388,608 bf16 each
    __bf16* Q   = (__bf16*)d_ws;
    __bf16* Kb  = Q  + QSZ;
    __bf16* Vt  = Kb + QSZ;                            // transposed [B,H,D,S]
    __bf16* ctx = Vt + QSZ;                            // 4*2048*1024 bf16

    dim3 blk(256);
    proj_kernel<false><<<dim3(8, 64, 1), blk, 0, stream>>>(query, Wq, bq, Q);
    proj_kernel<false><<<dim3(8, 64, 1), blk, 0, stream>>>(key_,  Wk, bk, Kb);
    proj_kernel<true ><<<dim3(8, 64, 1), blk, 0, stream>>>(value, Wv, bv, Vt);
    scores_kernel<<<dim3(16, 16, 32), blk, 0, stream>>>(Q, Kb, attn);
    softmax_kernel<<<dim3(65536, 1, 1), 256, 0, stream>>>(attn);
    attnv_kernel<<<dim3(1, 16, 32), blk, 0, stream>>>(attn, Vt, ctx);
    out_kernel<<<dim3(8, 64, 1), blk, 0, stream>>>(ctx, Wo, bo, xout);
}
